// WinTopKAttention_75874892251735
// MI455X (gfx1250) — matrix operand order repaired and checked by reading, not compiled, not run on hardware
//
#include <hip/hip_runtime.h>
#include <hip/hip_bf16.h>
#include <math.h>

typedef __attribute__((ext_vector_type(16))) _Float16 hv16;
typedef __attribute__((ext_vector_type(8)))  float    fv8;
typedef __attribute__((ext_vector_type(4)))  int      v4i;

#define NHEAD   8
#define HDIM    32
#define CEMB    256
#define S_WIN   64
#define TOPK    16
#define BNWIN   4608            // 2 * 2304 windows
#define MTOK    73728           // 4608 * 16 selected tokens
#define TOK_PER_BATCH 36864     // 2304 * 16
#define NWIN_SIDE 48            // 384 / 8

// ---------------------------------------------------------------- utilities

__device__ __forceinline__ float elu1(float x) {
    return x > 0.0f ? x + 1.0f : __expf(x);
}

__device__ __forceinline__ float block_reduce_sum_256(float v, float* red) {
    int tid = threadIdx.x;
    red[tid] = v;
    __syncthreads();
    for (int s = 128; s > 0; s >>= 1) {
        if (tid < s) red[tid] += red[tid + s];
        __syncthreads();
    }
    float r = red[0];
    __syncthreads();
    return r;
}

// ---- CDNA5 async global->LDS copy (ASYNCcnt path); builtin confirmed by probe.

__device__ __forceinline__ void gl_async_b128(const void* g, void* lds) {
#if __has_builtin(__builtin_amdgcn_global_load_async_to_lds_b128)
    __builtin_amdgcn_global_load_async_to_lds_b128(
        (v4i*)const_cast<void*>(g), (v4i*)lds, 0, 0);
#else
    unsigned off = (unsigned)(size_t)lds;
    asm volatile("global_load_async_to_lds_b128 %0, %1, off"
                 :: "v"(off), "v"(g) : "memory");
#endif
}

__device__ __forceinline__ void wait_async0() {
#if __has_builtin(__builtin_amdgcn_s_wait_asynccnt)
    __builtin_amdgcn_s_wait_asynccnt(0);
#else
    asm volatile("s_wait_asynccnt 0x0" ::: "memory");
#endif
}

__global__ void zero_f32_kernel(float* p, int n) {
    int i = blockIdx.x * blockDim.x + threadIdx.x;
    if (i < n) p[i] = 0.0f;
}

// dst[N][K] = (f16) src[K][N]  -- weights converted pre-transposed so the
// GEMM B-tile can be staged with contiguous async b128 copies.
__global__ void f32_to_f16_transpose_kernel(const float* __restrict__ src,
                                            _Float16* __restrict__ dst, int K, int N) {
    int i = blockIdx.x * blockDim.x + threadIdx.x;
    if (i < K * N) {
        int n = i / K, k = i % K;
        dst[i] = (_Float16)src[(size_t)k * N + n];
    }
}

// ------------------------------------------------------- top-k per window

__global__ __launch_bounds__(256) void topk_select_kernel(
        const float* __restrict__ X, _Float16* __restrict__ sel, int* __restrict__ idx_out) {
    __shared__ float sx[S_WIN * CEMB];   // 64 KB
    __shared__ float avg[CEMB];
    __shared__ float red[256];
    __shared__ float cosv[S_WIN];
    __shared__ int   selmap[TOPK];

    const int bn = blockIdx.x, tid = threadIdx.x;

    // async-stage the whole window (4096 x 16B chunks)
    for (int ch = tid; ch < 4096; ch += 256)
        gl_async_b128(X + (size_t)bn * (S_WIN * CEMB) + ch * 4, (char*)sx + ch * 16);
    wait_async0();
    __syncthreads();

    float s = 0.0f;
    for (int t = 0; t < S_WIN; ++t) s += sx[t * CEMB + tid];
    float a = s * (1.0f / 64.0f);
    avg[tid] = a;
    __syncthreads();

    float na2 = block_reduce_sum_256(a * a, red);
    float na = fmaxf(sqrtf(na2), 1e-8f);

    if (tid < S_WIN) {
        float dot = 0.0f, nb2 = 0.0f;
        for (int c = 0; c < CEMB; ++c) {
            float xv = sx[tid * CEMB + c];
            dot += avg[c] * xv;
            nb2 += xv * xv;
        }
        float nb = fmaxf(sqrtf(nb2), 1e-8f);
        cosv[tid] = dot / (na * nb);
    }
    __syncthreads();

    if (tid < S_WIN) {
        float mc = cosv[tid];
        int r = 0;
        for (int j = 0; j < S_WIN; ++j) {
            float cj = cosv[j];
            if (cj > mc || (cj == mc && j < tid)) ++r;
        }
        if (r < TOPK) {
            selmap[r] = tid;
            idx_out[bn * TOPK + r] = tid;
        }
    }
    __syncthreads();

    for (int i = tid; i < TOPK * CEMB; i += 256) {
        int r = i >> 8, c = i & 255;
        sel[((size_t)bn * TOPK + r) * CEMB + c] = (_Float16)sx[selmap[r] * CEMB + c];
    }
}

// ------------------------------------------------------------ WMMA GEMM
// C[M,N] = A[M,K] * B[K,N]; A f16 [M,K] row-major, B given PRE-TRANSPOSED as
// Bt f16 [N,K]. f32 accumulate. Block: 256 threads = 8 waves; tile 128x128;
// K-step 64 (2 x 16x16x32 sub-steps) -> 16 WMMA per barrier.
// Both tiles staged by global_load_async_to_lds_b128 into double-buffered LDS
// (2 x 32 KB); next tile's DMA overlaps the current tile's WMMA chain.
// B fragments flow through a 1-deep rotating pipeline: loads for fragment nt+1
// are issued before the WMMA consuming fragment nt, so waitcnt insertion can
// use partial dscnt waits and overlap LDS latency with the matrix pipe.

__global__ __launch_bounds__(256) void gemm_wmma_f16_kernel(
        const _Float16* __restrict__ A, const _Float16* __restrict__ Bt,
        float* __restrict__ Cf, _Float16* __restrict__ Ch,
        int M, int N, int K, int relu_f16_out) {
    __shared__ _Float16 smem[2 * 16384];  // per buffer: As[128*64] | Bs[128*64]

    const int tid  = threadIdx.x;
    const int wave = tid >> 5;
    const int lane = tid & 31;
    const int mblock = blockIdx.x * 128;
    const int nblock = blockIdx.y * 128;

    fv8 acc[8] = {{}, {}, {}, {}, {}, {}, {}, {}};

    const int l15 = lane & 15;
    const int akb = (lane < 16) ? 0 : 16;   // A frag: lanes 16-31 hold K 16..31
    const int bkb = (lane < 16) ? 0 : 16;   // B frag: lanes 16-31 hold K 16..31

    // stage one 128x64 tile pair: 1024 x 16B chunks each (8 asyncs/thread)
    auto issue_tile = [&](int buf, int k0) {
        _Float16* As = smem + buf * 16384;
        _Float16* Bs = As + 8192;
#pragma unroll
        for (int cc = 0; cc < 4; ++cc) {
            int c = tid + cc * 256;
            int row = c >> 3, part = c & 7;
            gl_async_b128(A  + (size_t)(mblock + row) * K + k0 + part * 8,
                          As + row * 64 + part * 8);
            gl_async_b128(Bt + (size_t)(nblock + row) * K + k0 + part * 8,
                          Bs + row * 64 + part * 8);
        }
    };

    const int nsteps = K >> 6;   // K/64; K is 256 or 512
    issue_tile(0, 0);

    for (int s = 0; s < nsteps; ++s) {
        wait_async0();       // own async stores to LDS complete
        __syncthreads();     // tile s visible; all waves done reading buffer s^1
        if (s + 1 < nsteps) issue_tile((s + 1) & 1, (s + 1) << 6);

        const _Float16* As = smem + (s & 1) * 16384;
        const _Float16* Bs = As + 8192;

#pragma unroll
        for (int half = 0; half < 2; ++half) {
            const int ks = half * 32;

            // A fragment (ISA 7.12.2: 16-bit A 16x32)
            hv16 afrag;
            {
                const _Float16* ar = &As[(wave * 16 + l15) * 64 + ks];
#pragma unroll
                for (int i = 0; i < 4; ++i) {
                    afrag[2 * i]     = ar[akb + 2 * i];
                    afrag[2 * i + 1] = ar[akb + 2 * i + 1];
                }
#pragma unroll
                for (int i = 0; i < 4; ++i) {
                    afrag[8 + 2 * i] = ar[akb + 8 + 2 * i];
                    afrag[9 + 2 * i] = ar[akb + 9 + 2 * i];
                }
            }

            const _Float16* bbase = &Bs[l15 * 64 + ks + bkb];
            auto load_b = [&](hv16& dst, int nt) {
                const _Float16* br = bbase + nt * (16 * 64);
#pragma unroll
                for (int i = 0; i < 8; ++i) {
                    dst[2 * i]     = br[2 * i];
                    dst[2 * i + 1] = br[2 * i + 1];
                }
            };

            // 1-deep rotating pipeline over the 8 column tiles
            hv16 bcur, bnext;
            load_b(bcur, 0);
#pragma unroll
            for (int nt = 0; nt < 8; ++nt) {
                if (nt < 7) load_b(bnext, nt + 1);
                acc[nt] = __builtin_amdgcn_wmma_f32_16x16x32_f16(
                    false, afrag, false, bcur, (short)0, acc[nt], false, false);
                if (nt < 7) bcur = bnext;
            }
        }
        __syncthreads();     // all waves done with buffer s before it is re-filled
    }

    // C/D layout: VGPR r -> M = r + 8*(lane>=16); N = lane&15
    const int mh = (lane >> 4) & 1;
    for (int nt = 0; nt < 8; ++nt) {
#pragma unroll
        for (int r = 0; r < 8; ++r) {
            int rowg = mblock + wave * 16 + r + mh * 8;
            int colg = nblock + nt * 16 + l15;
            float v = acc[nt][r];
            if (relu_f16_out)
                Ch[(size_t)rowg * N + colg] = (_Float16)fmaxf(v, 0.0f);
            else
                Cf[(size_t)rowg * N + colg] = v;
        }
    }
}

// ----------------------------------------------- linear attention reductions
// KV[b,h,d,v] = sum_t elu1(k[t,h,d]) * v[t,h,v];  Ksum[b,h,d] = sum_t elu1(k)
// (the reference's v/L ... *L pair cancels exactly and is dropped)

__global__ __launch_bounds__(256) void kv_reduce_kernel(
        const float* __restrict__ kf, const float* __restrict__ vf,
        float* __restrict__ KV, float* __restrict__ Ksum) {
    __shared__ float kb[16 * CEMB];
    __shared__ float vb[16 * CEMB];
    const int tid = threadIdx.x;
    const int h = tid >> 5, d = tid & 31;
    const int t0 = blockIdx.x * 64;
    const int b = t0 / TOK_PER_BATCH;

    float acc[32];
#pragma unroll
    for (int j = 0; j < 32; ++j) acc[j] = 0.0f;
    float ks = 0.0f;

    for (int chunk = 0; chunk < 4; ++chunk) {
        int tc = t0 + chunk * 16;
        for (int ch = tid; ch < 1024; ch += 256) {  // 16*256 floats = 1024 x 16B
            gl_async_b128(kf + (size_t)tc * CEMB + ch * 4, (char*)kb + ch * 16);
            gl_async_b128(vf + (size_t)tc * CEMB + ch * 4, (char*)vb + ch * 16);
        }
        wait_async0();
        __syncthreads();
        for (int tt = 0; tt < 16; ++tt) {
            float kd = elu1(kb[tt * CEMB + h * HDIM + d]);
            ks += kd;
            const float* vrow = &vb[tt * CEMB + h * HDIM];
#pragma unroll
            for (int j = 0; j < 32; ++j) acc[j] += kd * vrow[j];
        }
        __syncthreads();
    }

    float* kvb = &KV[(((size_t)b * NHEAD + h) * HDIM + d) * HDIM];
#pragma unroll
    for (int j = 0; j < 32; ++j) atomicAdd(&kvb[j], acc[j]);
    atomicAdd(&Ksum[((size_t)b * NHEAD + h) * HDIM + d], ks);
}

__global__ __launch_bounds__(256) void attn_apply_kernel(
        const float* __restrict__ qf, const float* __restrict__ KV,
        const float* __restrict__ Ksum, _Float16* __restrict__ attn_h) {
    __shared__ float kvl[NHEAD * HDIM * HDIM];  // 32 KB
    __shared__ float ksl[NHEAD * HDIM];
    const int tid = threadIdx.x;
    const int t0 = blockIdx.x * 32;
    const int b = t0 / TOK_PER_BATCH;

    for (int ch = tid; ch < 2048; ch += 256)    // 8192 floats = 2048 x 16B
        gl_async_b128(KV + (size_t)b * 8192 + ch * 4, (char*)kvl + ch * 16);
    ksl[tid] = Ksum[(size_t)b * 256 + tid];
    wait_async0();
    __syncthreads();

    const int lt = tid >> 3, h = tid & 7;
    const int t = t0 + lt;

    float Q[32];
    float zden = 0.0f;
#pragma unroll
    for (int dd = 0; dd < 32; ++dd) {
        float qq = elu1(qf[(size_t)t * CEMB + h * HDIM + dd]);
        Q[dd] = qq;
        zden += qq * ksl[h * HDIM + dd];
    }
    float z = 1.0f / (zden + 1e-6f);

    for (int jv = 0; jv < 32; ++jv) {
        float o = 0.0f;
#pragma unroll
        for (int dd = 0; dd < 32; ++dd) o += Q[dd] * kvl[(h * HDIM + dd) * HDIM + jv];
        attn_h[(size_t)t * CEMB + h * HDIM + jv] = (_Float16)(o * z);
    }
}

// ------------------------------------------------- LayerNorms / concat / res

__global__ __launch_bounds__(256) void ln1_concat_kernel(
        const float* __restrict__ msg, const _Float16* __restrict__ sel1h,
        const float* __restrict__ w, const float* __restrict__ bb,
        _Float16* __restrict__ cat) {
    __shared__ float red[256];
    const int t = blockIdx.x, c = threadIdx.x;
    float x = msg[(size_t)t * CEMB + c];
    float mean = block_reduce_sum_256(x, red) * (1.0f / 256.0f);
    float d = x - mean;
    float var = block_reduce_sum_256(d * d, red) * (1.0f / 256.0f);
    float y = d * rsqrtf(var + 1e-5f) * w[c] + bb[c];
    cat[(size_t)t * 512 + c]       = sel1h[(size_t)t * CEMB + c];
    cat[(size_t)t * 512 + 256 + c] = (_Float16)y;
}

__global__ __launch_bounds__(256) void ln2_residual_kernel(
        const float* __restrict__ msg2, const _Float16* __restrict__ sel1h,
        const float* __restrict__ w, const float* __restrict__ bb,
        float* __restrict__ out_tok) {
    __shared__ float red[256];
    const int t = blockIdx.x, c = threadIdx.x;
    float x = msg2[(size_t)t * CEMB + c];
    float mean = block_reduce_sum_256(x, red) * (1.0f / 256.0f);
    float d = x - mean;
    float var = block_reduce_sum_256(d * d, red) * (1.0f / 256.0f);
    float y = d * rsqrtf(var + 1e-5f) * w[c] + bb[c];
    out_tok[(size_t)t * CEMB + c] = (float)sel1h[(size_t)t * CEMB + c] + y;
}

// ------------------------------------- scatter_add + window_reverse (NCHW)

__global__ __launch_bounds__(256) void combine_kernel(
        const float* __restrict__ X, const int* __restrict__ idx,
        const float* __restrict__ out_tok, float* __restrict__ out) {
    __shared__ int map[S_WIN];
    const int bn = blockIdx.x, tid = threadIdx.x;
    if (tid < S_WIN) map[tid] = -1;
    __syncthreads();
    if (tid < TOPK) {
        int s = idx[bn * TOPK + tid];
        map[s] = tid;
    }
    __syncthreads();

    const int b = bn / 2304, win = bn % 2304;
    const int wr = win / NWIN_SIDE, wc = win % NWIN_SIDE;

    for (int i = tid; i < S_WIN * CEMB; i += 256) {
        int s = i >> 8;
        int c = i & 255;
        float v = X[(size_t)bn * (S_WIN * CEMB) + i];
        int r = map[s];
        if (r >= 0) v += out_tok[(((size_t)bn * TOPK + r) << 8) + c];
        int rr = s >> 3, cc = s & 7;
        int hh = wr * 8 + rr, ww = wc * 8 + cc;
        out[(((size_t)b * CEMB + c) * 384 + hh) * 384 + ww] = v;
    }
}

// ---------------------------------------------------------------- launcher

extern "C" void kernel_launch(void* const* d_in, const int* in_sizes, int n_in,
                              void* d_out, int out_size, void* d_ws, size_t ws_size,
                              hipStream_t stream) {
    (void)in_sizes; (void)n_in; (void)out_size; (void)ws_size;

    const float* x    = (const float*)d_in[0];
    const float* ssrc = (const float*)d_in[1];
    const float* Wq   = (const float*)d_in[2];
    const float* Wk   = (const float*)d_in[3];
    const float* Wv   = (const float*)d_in[4];
    const float* Wm   = (const float*)d_in[5];
    const float* W1   = (const float*)d_in[6];
    const float* W2   = (const float*)d_in[7];
    const float* n1w  = (const float*)d_in[8];
    const float* n1b  = (const float*)d_in[9];
    const float* n2w  = (const float*)d_in[10];
    const float* n2b  = (const float*)d_in[11];
    float* out = (float*)d_out;

    char* ws = (char*)d_ws;
    size_t off = 0;
    auto alloc = [&](size_t bytes) -> void* {
        void* p = ws + off;
        off = (off + bytes + 255) & ~(size_t)255;
        return p;
    };

    _Float16* sel1h = (_Float16*)alloc((size_t)MTOK * CEMB * 2);
    _Float16* sel2h = (_Float16*)alloc((size_t)MTOK * CEMB * 2);
    int* idx1 = (int*)alloc((size_t)BNWIN * TOPK * 4);
    int* idx2 = (int*)alloc((size_t)BNWIN * TOPK * 4);
    _Float16* wqt = (_Float16*)alloc(65536 * 2);    // [N=256][K=256]
    _Float16* wkt = (_Float16*)alloc(65536 * 2);
    _Float16* wvt = (_Float16*)alloc(65536 * 2);
    _Float16* wmt = (_Float16*)alloc(65536 * 2);
    _Float16* w1t = (_Float16*)alloc(262144 * 2);   // [N=512][K=512]
    _Float16* w2t = (_Float16*)alloc(131072 * 2);   // [N=256][K=512]
    float* qf   = (float*)alloc((size_t)MTOK * CEMB * 4);
    float* kf   = (float*)alloc((size_t)MTOK * CEMB * 4);
    float* vf   = (float*)alloc((size_t)MTOK * CEMB * 4);
    float* KV   = (float*)alloc(2 * NHEAD * HDIM * HDIM * 4);
    float* Ksum = (float*)alloc(2 * NHEAD * HDIM * 4);
    _Float16* attnh = (_Float16*)alloc((size_t)MTOK * CEMB * 2);
    float* msgf  = (float*)alloc((size_t)MTOK * CEMB * 4);
    _Float16* cath = (_Float16*)alloc((size_t)MTOK * 512 * 2);
    _Float16* hidh = (_Float16*)alloc((size_t)MTOK * 512 * 2);
    float* msg2f  = (float*)alloc((size_t)MTOK * CEMB * 4);
    float* outtok = (float*)alloc((size_t)MTOK * CEMB * 4);

    // zero the attention accumulators (atomics target)
    zero_f32_kernel<<<(2 * NHEAD * HDIM * HDIM + 255) / 256, 256, 0, stream>>>(KV, 2 * NHEAD * HDIM * HDIM);
    zero_f32_kernel<<<(2 * NHEAD * HDIM + 255) / 256, 256, 0, stream>>>(Ksum, 2 * NHEAD * HDIM);

    // weight conversion to f16, pre-transposed to [N][K]
    f32_to_f16_transpose_kernel<<<65536 / 256, 256, 0, stream>>>(Wq, wqt, 256, 256);
    f32_to_f16_transpose_kernel<<<65536 / 256, 256, 0, stream>>>(Wk, wkt, 256, 256);
    f32_to_f16_transpose_kernel<<<65536 / 256, 256, 0, stream>>>(Wv, wvt, 256, 256);
    f32_to_f16_transpose_kernel<<<65536 / 256, 256, 0, stream>>>(Wm, wmt, 256, 256);
    f32_to_f16_transpose_kernel<<<262144 / 256, 256, 0, stream>>>(W1, w1t, 512, 512);
    f32_to_f16_transpose_kernel<<<131072 / 256, 256, 0, stream>>>(W2, w2t, 512, 256);

    // top-k selection
    topk_select_kernel<<<BNWIN, 256, 0, stream>>>(x, sel1h, idx1);
    topk_select_kernel<<<BNWIN, 256, 0, stream>>>(ssrc, sel2h, idx2);

    // q/k/v projections (WMMA)
    dim3 g256(MTOK / 128, 256 / 128);
    dim3 g512(MTOK / 128, 512 / 128);
    gemm_wmma_f16_kernel<<<g256, 256, 0, stream>>>(sel1h, wqt, qf, nullptr, MTOK, 256, 256, 0);
    gemm_wmma_f16_kernel<<<g256, 256, 0, stream>>>(sel2h, wkt, kf, nullptr, MTOK, 256, 256, 0);
    gemm_wmma_f16_kernel<<<g256, 256, 0, stream>>>(sel2h, wvt, vf, nullptr, MTOK, 256, 256, 0);

    // linear attention
    kv_reduce_kernel<<<MTOK / 64, 256, 0, stream>>>(kf, vf, KV, Ksum);
    attn_apply_kernel<<<MTOK / 32, 256, 0, stream>>>(qf, KV, Ksum, attnh);

    // merge projection + LN1 + concat
    gemm_wmma_f16_kernel<<<g256, 256, 0, stream>>>(attnh, wmt, msgf, nullptr, MTOK, 256, 256, 0);
    ln1_concat_kernel<<<MTOK, 256, 0, stream>>>(msgf, sel1h, n1w, n1b, cath);

    // MLP: relu(cat @ W1) @ W2, then LN2 + residual
    gemm_wmma_f16_kernel<<<g512, 256, 0, stream>>>(cath, w1t, nullptr, hidh, MTOK, 512, 512, 1);
    gemm_wmma_f16_kernel<<<g256, 256, 0, stream>>>(hidh, w2t, msg2f, nullptr, MTOK, 256, 512, 0);
    ln2_residual_kernel<<<MTOK, 256, 0, stream>>>(msg2f, sel1h, n2w, n2b, outtok);

    // scatter_add + window_reverse to NCHW output
    combine_kernel<<<BNWIN, 256, 0, stream>>>(x, idx1, outtok, out);
}
